// LSTM_u_80642305950310
// MI455X (gfx1250) — matrix operand order repaired and checked by reading, not compiled, not run on hardware
//
#include <hip/hip_runtime.h>

// Problem constants (from reference): N=256, T=512, E=256, H=256
#define N_  256
#define T_  512
#define E_  256
#define H_  256
#define G4  1024   // 4*H gate columns
#define KC  512    // concat K = E + H

typedef __attribute__((ext_vector_type(16))) __bf16 v16bf;
typedef __attribute__((ext_vector_type(8)))  __bf16 v8bf;
typedef __attribute__((ext_vector_type(8)))  float  v8f;
typedef __attribute__((ext_vector_type(4)))  float  v4f;

// ---------------------------------------------------------------------------
// Prep: pack Wcat[g][k] = (k<E ? W_ih[g][k] : W_hh[g][k-E]) as bf16, row-major.
// Row-major rows == gate columns == WMMA B columns, so B fragments are
// contiguous 32B per lane (no transpose needed).
// ---------------------------------------------------------------------------
__global__ __launch_bounds__(256) void lstm_prep_weights(
    const float* __restrict__ wih, const float* __restrict__ whh,
    __bf16* __restrict__ wcat)
{
  int idx  = blockIdx.x * blockDim.x + threadIdx.x;
  int base = idx * 8;                       // 8 elements per thread
  if (base >= G4 * KC) return;
  int g = base >> 9;                        // /KC
  int k = base & (KC - 1);
  const float* src = (k < E_) ? (wih + g * E_ + k) : (whh + g * H_ + (k - E_));
  v8bf o;
  #pragma unroll
  for (int i = 0; i < 8; ++i) o[i] = (__bf16)src[i];
  *(v8bf*)(wcat + base) = o;
}

__device__ __forceinline__ float fsigmoid(float x) {
  return 1.0f / (1.0f + __expf(-x));
}
__device__ __forceinline__ float ftanh(float x) {
  // 2*sigmoid(2x)-1 : saturates correctly at +/-1, one v_exp_f32
  return 2.0f / (1.0f + __expf(-2.0f * x)) - 1.0f;
}

// ---------------------------------------------------------------------------
// Fused LSTM scan. Grid: N/16 workgroups of 512 threads (16 waves).
// WG b handles samples [16b, 16b+16). Wave w owns h-columns [16w,16w+16)
// for all 4 gates (4 accumulator tiles). c,h persist in VGPRs; h is mirrored
// into LDS as bf16 each step to form next step's A-matrix rows [x_t | h].
// ---------------------------------------------------------------------------
__global__ __launch_bounds__(512) void lstm_scan(
    const float* __restrict__ x,      // [N,T,E]
    const int*   __restrict__ lens,   // [N]
    const float* __restrict__ bih,    // [4H]
    const float* __restrict__ bhh,    // [4H]
    const __bf16* __restrict__ wcat,  // [4H][KC] bf16
    float* __restrict__ out)          // [N,H]
{
  __shared__ __align__(32) __bf16 abuf[16 * KC];   // rows=samples, cols=[x|h]
  __shared__ int lenbuf[16];

  const int tid  = threadIdx.x;
  const int lane = tid & 31;
  const int wave = tid >> 5;
  const int n0   = blockIdx.x * 16;
  const int half = lane >> 4;        // 0: lanes 0-15, 1: lanes 16-31
  const int lcol = lane & 15;

  if (tid < 16) lenbuf[tid] = lens[n0 + tid];
  // zero the h region of abuf (K = E..KC)
  for (int i = tid; i < 16 * H_; i += 512) {
    int rr = i >> 8, cc = i & (H_ - 1);
    abuf[rr * KC + E_ + cc] = (__bf16)0.0f;
  }
  __syncthreads();

  int tmax = 0;
  #pragma unroll
  for (int i = 0; i < 16; ++i) tmax = max(tmax, lenbuf[i]);

  // D-tile layout: element r of v8f -> sample row m = r + half*8, col = lcol
  int mylen[8];
  #pragma unroll
  for (int r = 0; r < 8; ++r) mylen[r] = lenbuf[r + half * 8];

  float bias[4];
  #pragma unroll
  for (int g = 0; g < 4; ++g) {
    int gc = g * H_ + wave * 16 + lcol;
    bias[g] = bih[gc] + bhh[gc];
  }

  // A fragment base: lane's sample row = lcol; contiguous K: half*16 .. +15
  const __bf16* arow = abuf + lcol * KC + half * 16;
  // Per-gate B row offset (in elements) within wcat
  size_t boff[4];
  #pragma unroll
  for (int g = 0; g < 4; ++g)
    boff[g] = (size_t)(g * H_ + wave * 16 + lcol) * KC + half * 16;

  // x staging: each thread owns one 8-float chunk of the 16x256 x_t tile
  const int xrow = tid >> 5;
  const int xchk = (tid & 31) * 8;
  const float* xsrc = x + (size_t)(n0 + xrow) * T_ * E_ + xchk;
  __bf16* xdst = abuf + xrow * KC + xchk;

  v8f hreg, creg;
  #pragma unroll
  for (int r = 0; r < 8; ++r) { hreg[r] = 0.0f; creg[r] = 0.0f; }

  int zoff = 0;   // always 0, but opaque to the optimizer (see asm below)

  for (int t = 0; t < tmax; ++t) {
    // Make the weight address formally loop-variant so the 64 B-fragment
    // loads are NOT hoisted out of the loop and spilled to scratch; they
    // must re-issue each step as global_load_b128 against L2-hot weights.
    asm volatile("" : "+s"(zoff));
    const __bf16* wbase = wcat + zoff;

    // ---- write phase: stage x_t (h was written at end of previous step) ----
    {
      const v4f* s = (const v4f*)(xsrc + (size_t)t * E_);
      v4f f0 = __builtin_nontemporal_load(s);      // stream x, keep L2 for W
      v4f f1 = __builtin_nontemporal_load(s + 1);
      v8bf xa;
      xa[0] = (__bf16)f0.x; xa[1] = (__bf16)f0.y;
      xa[2] = (__bf16)f0.z; xa[3] = (__bf16)f0.w;
      xa[4] = (__bf16)f1.x; xa[5] = (__bf16)f1.y;
      xa[6] = (__bf16)f1.z; xa[7] = (__bf16)f1.w;
      *(v8bf*)xdst = xa;
    }
    __syncthreads();

    // ---- GEMM phase: gates(16x64 this wave) = [x_t|h](16xKC) @ Wcat^T ----
    v8f acc[4] = {};   // zero C -> first WMMA can use inline-0 C operand

    #pragma unroll
    for (int kf = 0; kf < 16; ++kf) {
      const int kb = kf * 32;
      v16bf a = *(const v16bf*)(arow + kb);        // K: kb + half*16 .. +15
      #pragma unroll
      for (int g = 0; g < 4; ++g) {
        v16bf b = *(const v16bf*)(wbase + boff[g] + kb);  // 32B, L2-hot
        acc[g] = __builtin_amdgcn_wmma_f32_16x16x32_bf16(
            false, a, false, b, (short)0, acc[g], false, false);
      }
    }
    __syncthreads();

    // ---- elementwise phase: all 4 gates for (sample m, h-col) in-lane ----
    #pragma unroll
    for (int r = 0; r < 8; ++r) {
      float iv = fsigmoid(acc[0][r] + bias[0]);
      float fv = fsigmoid(acc[1][r] + bias[1]);
      float gv = ftanh   (acc[2][r] + bias[2]);
      float ov = fsigmoid(acc[3][r] + bias[3]);
      float cn = fv * creg[r] + iv * gv;
      float hn = ov * ftanh(cn);
      bool valid = t < mylen[r];
      creg[r] = valid ? cn : creg[r];
      hreg[r] = valid ? hn : hreg[r];
      abuf[(r + half * 8) * KC + E_ + wave * 16 + lcol] = (__bf16)hreg[r];
    }
  }

  #pragma unroll
  for (int r = 0; r < 8; ++r)
    out[(size_t)(n0 + r + half * 8) * H_ + wave * 16 + lcol] = hreg[r];
}

// ---------------------------------------------------------------------------
extern "C" void kernel_launch(void* const* d_in, const int* in_sizes, int n_in,
                              void* d_out, int out_size, void* d_ws, size_t ws_size,
                              hipStream_t stream) {
  const float* x   = (const float*)d_in[0];   // self_attn [N,T,E]
  const int*   tl  = (const int*)  d_in[1];   // traj_len  [N]
  const float* wih = (const float*)d_in[2];   // W_ih [4H,E]
  const float* whh = (const float*)d_in[3];   // W_hh [4H,H]
  const float* bih = (const float*)d_in[4];
  const float* bhh = (const float*)d_in[5];
  __bf16* wcat = (__bf16*)d_ws;               // 4H*KC*2 = 1 MB scratch
  float*  out  = (float*)d_out;               // [N,1,H]

  const int prep_elems = (G4 * KC) / 8;       // 65536 threads
  lstm_prep_weights<<<prep_elems / 256, 256, 0, stream>>>(wih, whh, wcat);
  lstm_scan<<<N_ / 16, 512, 0, stream>>>(x, tl, bih, bhh, wcat, out);
}
